// Model_15917148799899
// MI455X (gfx1250) — matrix operand order repaired and checked by reading, not compiled, not run on hardware
//
#include <hip/hip_runtime.h>

// ---------------------------------------------------------------------------
// Quaternion routing layer for MI455X (gfx1250, wave32, WMMA bf16 16x16x32).
// Heavy GEMMs run as split-precision bf16 (hi+lo) WMMA with f32 accumulation:
//   acc += A_hi*B_hi + A_hi*B_lo + A_lo*B_hi   (~2^-16 rel err vs fp32)
// B fragments are streamed into double-buffered LDS with
// GLOBAL_LOAD_ASYNC_TO_LDS_B128 (ASYNCcnt), overlapping DMA with WMMA.
// Fragment layouts per ISA 7.12.2:
//   A (16x32 bf16): lane = h*16+m,  K = 16*(v>>2) + 8*h + 2*(v&3) + e
//   B (32x16 bf16): lane = h*16+n,  K = 16*(v>>2) + 8*h + 2*(v&3) + e
// ---------------------------------------------------------------------------

typedef __attribute__((ext_vector_type(16))) __bf16 bf16x16;
typedef __attribute__((ext_vector_type(8)))  float  floatx8;

#define D_DIM 512
#define QD    128
#define KCODE 64
#define NTOK  4096
#define BATCH 8
#define MSLOT 3
#define ROWS_BIG   (BATCH * NTOK)        // 32768
#define RT_BIG     (ROWS_BIG / 16)       // 2048 row tiles
#define ROWS_PROTO (BATCH * KCODE)       // 512
#define RT_PROTO   (ROWS_PROTO / 16)     // 32 row tiles
#define KSTEPS     (D_DIM / 32)          // 16
#define STAGE_BYTES 65536                // per k-step: 32KB hi + 32KB lo

// quaternion block structure: W[rc*128+a, cc*128+b] = sign * w4[comp][a][b]
__device__ __constant__ int   CTAB[16] = {0,1,2,3, 1,0,3,2, 2,3,0,1, 3,2,1,0};
__device__ __constant__ float STAB[16] = {1.f,-1.f,-1.f,-1.f,
                                          1.f, 1.f,-1.f, 1.f,
                                          1.f, 1.f, 1.f,-1.f,
                                          1.f,-1.f, 1.f, 1.f};

__device__ __forceinline__ unsigned bpack(__bf16 a, __bf16 b) {
  unsigned short ua, ub;
  __builtin_memcpy(&ua, &a, 2);
  __builtin_memcpy(&ub, &b, 2);
  return (unsigned)ua | ((unsigned)ub << 16);
}

// split (x0,x1) into bf16 hi pair + bf16 lo (residual) pair, packed as u32
__device__ __forceinline__ void split_pair(float x0, float x1,
                                           unsigned& hi, unsigned& lo) {
  __bf16 h0 = (__bf16)x0, h1 = (__bf16)x1;
  __bf16 l0 = (__bf16)(x0 - (float)h0);
  __bf16 l1 = (__bf16)(x1 - (float)h1);
  hi = bpack(h0, h1);
  lo = bpack(l0, l1);
}

// Store one adjacent-k pair (k even) of an A fragment (hi/lo split).
__device__ __forceinline__ void store_afrag_pair(unsigned* __restrict__ afHi,
                                                 unsigned* __restrict__ afLo,
                                                 int rtile, int m, int k,
                                                 float x0, float x1) {
  int ks  = k >> 5;
  int k32 = k & 31;
  int h   = (k32 >> 3) & 1;
  int v   = ((k32 >> 4) & 1) * 4 + ((k32 >> 1) & 3);
  int lane = h * 16 + m;
  size_t idx = ((size_t)(rtile * 16 + ks) * 32 + lane) * 8 + v; // u32 units
  unsigned hi, lo;
  split_pair(x0, x1, hi, lo);
  afHi[idx] = hi;
  afLo[idx] = lo;
}

// ---- CDNA5 async global->LDS DMA (ASYNCcnt) -------------------------------
// VDST VGPR = per-lane LDS byte address (allocation-relative; low 32 bits of
// a generic LDS pointer per ISA 10.2), VADDR = 64-bit global address.
__device__ __forceinline__ void async_b128_to_lds(void* lds_generic,
                                                  const void* gptr) {
  unsigned lds_off = (unsigned)(unsigned long long)(uintptr_t)lds_generic;
  unsigned long long ga = (unsigned long long)(uintptr_t)gptr;
  asm volatile("global_load_async_to_lds_b128 %0, %1, off"
               :: "v"(lds_off), "v"(ga)
               : "memory");
}

__device__ __forceinline__ void wait_async0() {
#if __has_builtin(__builtin_amdgcn_s_wait_asynccnt)
  __builtin_amdgcn_s_wait_asynccnt(0);
#else
  asm volatile("s_wait_asynccnt 0x0" ::: "memory");
#endif
}

// ---------------------------------------------------------------------------
// Kernel 1: segmented weighted sum into proto_raw / wsum.
// 128 KB LDS accumulator per block (fits 320 KB WGP LDS), ds_add_f32 atomics.
// ---------------------------------------------------------------------------
__global__ __launch_bounds__(256)
void segsum_kernel(const float* __restrict__ q, const int* __restrict__ aidx,
                   const float* __restrict__ aw, const float* __restrict__ cmask,
                   float* __restrict__ protoRaw, float* __restrict__ wsum) {
  extern __shared__ float lds[];                 // 64*512 + 64 floats
  float* lp = lds;
  float* lw = lds + KCODE * D_DIM;
  const int tid   = threadIdx.x;
  const int b     = blockIdx.x >> 5;             // 32 chunks per batch
  const int chunk = blockIdx.x & 31;

  for (int i = tid; i < KCODE * D_DIM; i += 256) lp[i] = 0.f;
  if (tid < KCODE) lw[tid] = 0.f;
  __syncthreads();

  const int sub = tid >> 7;                      // which of 2 tokens
  const int dl  = tid & 127;                     // float4 lane over 512 dims
  const float4* q4 = (const float4*)q;

  for (int off = 0; off < 128; off += 2) {
    int n   = chunk * 128 + off + sub;
    int row = b * NTOK + n;
    float msk = cmask[row];
    float4 qv = q4[(size_t)row * 128 + dl];
    #pragma unroll
    for (int s = 0; s < MSLOT; ++s) {
      int ii   = row * MSLOT + s;
      int code = aidx[ii];
      float w  = aw[ii] * msk;
      float* dst = lp + code * D_DIM + dl * 4;
      atomicAdd(dst + 0, qv.x * w);
      atomicAdd(dst + 1, qv.y * w);
      atomicAdd(dst + 2, qv.z * w);
      atomicAdd(dst + 3, qv.w * w);
      if (dl == 0) atomicAdd(lw + code, w);
    }
  }
  __syncthreads();

  for (int i = tid; i < KCODE * D_DIM; i += 256)
    atomicAdd(protoRaw + (size_t)b * (KCODE * D_DIM) + i, lp[i]);
  if (tid < KCODE)
    atomicAdd(wsum + b * KCODE + tid, lw[tid]);
}

// ---------------------------------------------------------------------------
// Kernel 2: pack quaternion-structured W into WMMA B-fragment layout (hi/lo).
// ---------------------------------------------------------------------------
__global__ __launch_bounds__(256)
void weight_pack_kernel(const float* __restrict__ w4,
                        unsigned* __restrict__ bfHi,
                        unsigned* __restrict__ bfLo) {
  int t = blockIdx.x * 256 + threadIdx.x;        // 32*16*32 = 16384 tasks
  int ct   = t >> 9;
  int rem  = t & 511;
  int ks   = rem >> 5;
  int lane = rem & 31;
  int h    = lane >> 4;
  int n    = lane & 15;
  int oidx = ct * 16 + n;                        // output (column) index
  int rc   = oidx >> 7;
  int a    = oidx & 127;
  #pragma unroll
  for (int v = 0; v < 8; ++v) {
    int k0 = ks * 32 + 16 * (v >> 2) + 8 * h + 2 * (v & 3); // B layout: K = 16*(v>>2) + 8h + 2*(v&3) + e
    int cc = k0 >> 7;
    int bb = k0 & 127;
    int comp = CTAB[rc * 4 + cc];
    float sg = STAB[rc * 4 + cc];
    const float* blk = w4 + ((size_t)comp * QD + a) * QD;
    float w0 = sg * blk[bb];
    float w1 = sg * blk[bb + 1];
    unsigned hi, lo;
    split_pair(w0, w1, hi, lo);
    size_t idx = ((size_t)(ct * 16 + ks) * 32 + lane) * 8 + v;
    bfHi[idx] = hi;
    bfLo[idx] = lo;
  }
}

// ---------------------------------------------------------------------------
// Kernel 3: normalize proto_raw and pack into A fragments (hi/lo).
// ---------------------------------------------------------------------------
__global__ __launch_bounds__(256)
void proto_pack_kernel(const float* __restrict__ protoRaw,
                       const float* __restrict__ wsum,
                       unsigned* __restrict__ afHi,
                       unsigned* __restrict__ afLo) {
  int rt  = blockIdx.x;                          // 0..31
  int tid = threadIdx.x;
  for (int it = 0; it < 16; ++it) {
    int task = tid + 256 * it;                   // 16 rows * 256 k-pairs
    int m = task >> 8;
    int j = task & 255;
    int prow = rt * 16 + m;
    float inv = 1.f / (wsum[prow] + 1e-6f);
    float x0 = protoRaw[(size_t)prow * D_DIM + 2 * j] * inv;
    float x1 = protoRaw[(size_t)prow * D_DIM + 2 * j + 1] * inv;
    store_afrag_pair(afHi, afLo, rt, m, 2 * j, x0, x1);
  }
}

// ---------------------------------------------------------------------------
// Kernel 4: gather prototypes, Hamilton product, weighted sum over slots,
// pack msg directly into A fragments (hi/lo).
// ---------------------------------------------------------------------------
__global__ __launch_bounds__(256)
void msg_pack_kernel(const float* __restrict__ q, const int* __restrict__ aidx,
                     const float* __restrict__ aw,
                     const float* __restrict__ proto,  // post-linear (B,64,512)
                     unsigned* __restrict__ afHi, unsigned* __restrict__ afLo) {
  int rt  = blockIdx.x;                          // 0..2047
  int tid = threadIdx.x;
  const float2* q2 = (const float2*)q;
  const float2* p2 = (const float2*)proto;
  for (int it = 0; it < 4; ++it) {
    int task = tid + 256 * it;                   // 16 rows * 64 dd-pairs
    int m = task >> 6;
    int j = task & 63;                           // dd = 2*j in [0,128)
    int row = rt * 16 + m;
    int b   = row >> 12;                         // row / 4096
    size_t qb = (size_t)row * 256;               // float2 units per row

    float2 tr = q2[qb + j];
    float2 ti = q2[qb + 64 + j];
    float2 tj = q2[qb + 128 + j];
    float2 tk = q2[qb + 192 + j];

    float mrx = 0.f, mry = 0.f, mix = 0.f, miy = 0.f;
    float mjx = 0.f, mjy = 0.f, mkx = 0.f, mky = 0.f;

    #pragma unroll
    for (int s = 0; s < MSLOT; ++s) {
      int ii   = row * MSLOT + s;
      int code = aidx[ii];
      float w  = aw[ii];                          // msg uses assign_w (no mask)
      size_t pb = ((size_t)(b * KCODE + code)) * 256;
      float2 pr = p2[pb + j];
      float2 pi = p2[pb + 64 + j];
      float2 pj = p2[pb + 128 + j];
      float2 pk = p2[pb + 192 + j];
      // hamilton(token, proto)
      mrx += w * (tr.x * pr.x - ti.x * pi.x - tj.x * pj.x - tk.x * pk.x);
      mry += w * (tr.y * pr.y - ti.y * pi.y - tj.y * pj.y - tk.y * pk.y);
      mix += w * (tr.x * pi.x + ti.x * pr.x + tj.x * pk.x - tk.x * pj.x);
      miy += w * (tr.y * pi.y + ti.y * pr.y + tj.y * pk.y - tk.y * pj.y);
      mjx += w * (tr.x * pj.x - ti.x * pk.x + tj.x * pr.x + tk.x * pi.x);
      mjy += w * (tr.y * pj.y - ti.y * pk.y + tj.y * pr.y + tk.y * pi.y);
      mkx += w * (tr.x * pk.x + ti.x * pj.x - tj.x * pi.x + tk.x * pr.x);
      mky += w * (tr.y * pk.y + ti.y * pj.y - tj.y * pi.y + tk.y * pr.y);
    }
    int dd = 2 * j;
    store_afrag_pair(afHi, afLo, rt, m, 0 * QD + dd, mrx, mry);
    store_afrag_pair(afHi, afLo, rt, m, 1 * QD + dd, mix, miy);
    store_afrag_pair(afHi, afLo, rt, m, 2 * QD + dd, mjx, mjy);
    store_afrag_pair(afHi, afLo, rt, m, 3 * QD + dd, mkx, mky);
  }
}

// ---------------------------------------------------------------------------
// Kernel 5: split-precision bf16 WMMA GEMM, out = X @ W^T (+bias [+q, LN]).
// Block: 256 threads = 8 wave32. Wave (r2, comp): r2 selects one of 2 row
// tiles, comp selects a quaternion component (128 cols = 8 WMMA tiles).
// B fragments are double-buffered in LDS via async global->LDS DMA so the
// next k-step's 64 KB slice streams in while WMMA consumes the current one.
// Epilogue 1 fuses bias + residual + per-component LayerNorm.
// ---------------------------------------------------------------------------
__global__ __launch_bounds__(256)
void qgemm_kernel(const unsigned short* __restrict__ aHi,
                  const unsigned short* __restrict__ aLo,
                  const unsigned short* __restrict__ bHi,
                  const unsigned short* __restrict__ bLo,
                  const float* __restrict__ bias,
                  const float* __restrict__ qres,
                  const float* __restrict__ gamma,
                  const float* __restrict__ beta,
                  float* __restrict__ out, int epi) {
  extern __shared__ unsigned char smem[];        // 2 x 64 KB double buffer

  const int tid  = threadIdx.x;
  const int lane = tid & 31;
  const int wid  = tid >> 5;
  const int comp = wid & 3;
  const int r2   = wid >> 2;
  const int rtile = blockIdx.x * 2 + r2;

  floatx8 acc[8];
  floatx8 zero = {};
  #pragma unroll
  for (int t = 0; t < 8; ++t) acc[t] = zero;

  const uint4* gHi = (const uint4*)bHi;
  const uint4* gLo = (const uint4*)bLo;

  // issue one k-step's B-fragment stage (32 KB hi + 32 KB lo) as async DMA
  auto stage = [&](int ks, int buf) {
    unsigned char* base = smem + buf * STAGE_BYTES;
    #pragma unroll
    for (int r = 0; r < 8; ++r) {
      int s   = tid + 256 * r;                   // 0..2047 uint4 slots
      int ct  = s >> 6;
      int rem = s & 63;
      int ln  = rem >> 1;
      int jj  = rem & 1;
      size_t g = ((size_t)(ct * 16 + ks) * 32 + ln) * 2 + jj;
      async_b128_to_lds(base + (size_t)s * 16, gHi + g);
      async_b128_to_lds(base + 32768 + (size_t)s * 16, gLo + g);
    }
  };

  stage(0, 0);
  wait_async0();
  __syncthreads();

  for (int ks = 0; ks < KSTEPS; ++ks) {
    const int cur = ks & 1;
    if (ks + 1 < KSTEPS) stage(ks + 1, cur ^ 1); // prefetch next slice

    const uint4* sHi = (const uint4*)(smem + cur * STAGE_BYTES);
    const uint4* sLo = (const uint4*)(smem + cur * STAGE_BYTES + 32768);

    const bf16x16 a_hi =
        *(const bf16x16*)(aHi + ((size_t)(rtile * 16 + ks) * 32 + lane) * 16);
    const bf16x16 a_lo =
        *(const bf16x16*)(aLo + ((size_t)(rtile * 16 + ks) * 32 + lane) * 16);

    #pragma unroll
    for (int t = 0; t < 8; ++t) {
      int ct = comp * 8 + t;
      bf16x16 b_hi = *(const bf16x16*)(sHi + (ct * 32 + lane) * 2);
      bf16x16 b_lo = *(const bf16x16*)(sLo + (ct * 32 + lane) * 2);
      acc[t] = __builtin_amdgcn_wmma_f32_16x16x32_bf16(
          false, a_hi, false, b_hi, (short)0, acc[t], false, false);
      acc[t] = __builtin_amdgcn_wmma_f32_16x16x32_bf16(
          false, a_hi, false, b_lo, (short)0, acc[t], false, false);
      acc[t] = __builtin_amdgcn_wmma_f32_16x16x32_bf16(
          false, a_lo, false, b_hi, (short)0, acc[t], false, false);
    }

    wait_async0();         // next slice landed in LDS (this wave's DMAs)
    __syncthreads();       // cross-wave: everyone done reading + staging
  }

  // Epilogue. C/D layout: lane in [0,16): col = lane, rows 0-7 in VGPR p;
  // lane in [16,32): col = lane-16, rows 8-15 in VGPR p.
  const int colLane = lane & 15;
  const int half    = lane >> 4;

  if (epi == 0) {
    #pragma unroll
    for (int t = 0; t < 8; ++t) {
      int col = comp * QD + t * 16 + colLane;
      float bs = bias[col];
      #pragma unroll
      for (int p = 0; p < 8; ++p) {
        int grow = rtile * 16 + p + 8 * half;
        out[(size_t)grow * D_DIM + col] = acc[t][p] + bs;
      }
    }
  } else {
    // x = gemm + bias + q ; then LN over this wave's 128 columns per row
    #pragma unroll
    for (int t = 0; t < 8; ++t) {
      int col = comp * QD + t * 16 + colLane;
      float bs = bias[col];
      #pragma unroll
      for (int p = 0; p < 8; ++p) {
        int grow = rtile * 16 + p + 8 * half;
        acc[t][p] += bs + qres[(size_t)grow * D_DIM + col];
      }
    }
    float mu[8], rs[8];
    #pragma unroll
    for (int p = 0; p < 8; ++p) {
      float s1 = 0.f, s2 = 0.f;
      #pragma unroll
      for (int t = 0; t < 8; ++t) { float x = acc[t][p]; s1 += x; s2 += x * x; }
      // butterfly over the 16 lanes of this half (xor<=8 stays in half)
      for (int off = 1; off < 16; off <<= 1) {
        s1 += __shfl_xor(s1, off, 32);
        s2 += __shfl_xor(s2, off, 32);
      }
      float mean = s1 * (1.f / 128.f);
      float var  = s2 * (1.f / 128.f) - mean * mean;
      mu[p] = mean;
      rs[p] = rsqrtf(var + 1e-5f);
    }
    #pragma unroll
    for (int t = 0; t < 8; ++t) {
      int dcomp = t * 16 + colLane;
      int col   = comp * QD + dcomp;
      float g  = gamma[comp * QD + dcomp];
      float be = beta[comp * QD + dcomp];
      #pragma unroll
      for (int p = 0; p < 8; ++p) {
        int grow = rtile * 16 + p + 8 * half;
        out[(size_t)grow * D_DIM + col] = (acc[t][p] - mu[p]) * rs[p] * g + be;
      }
    }
  }
}

// ---------------------------------------------------------------------------
extern "C" void kernel_launch(void* const* d_in, const int* in_sizes, int n_in,
                              void* d_out, int out_size, void* d_ws,
                              size_t ws_size, hipStream_t stream) {
  (void)in_sizes; (void)n_in; (void)out_size; (void)ws_size;

  const float* q       = (const float*)d_in[0];
  const int*   aidx    = (const int*)  d_in[1];
  const float* aw      = (const float*)d_in[2];
  const float* cmask   = (const float*)d_in[3];
  const float* proto_w = (const float*)d_in[4];
  const float* proto_b = (const float*)d_in[5];
  const float* upd_w   = (const float*)d_in[6];
  const float* upd_b   = (const float*)d_in[7];
  const float* gamma   = (const float*)d_in[8];
  const float* beta    = (const float*)d_in[9];

  float* qnew      = (float*)d_out;                              // (8,4096,512)
  float* proto_out = (float*)d_out + (size_t)ROWS_BIG * D_DIM;   // (8,64,512)

  // workspace carve-up (~71 MB total)
  char* ws = (char*)d_ws;
  size_t off = 0;
  auto carve = [&](size_t bytes) -> void* {
    void* p = ws + off;
    off += (bytes + 255) & ~(size_t)255;
    return p;
  };
  float* protoRaw = (float*)carve((size_t)BATCH * KCODE * D_DIM * 4);  // 1 MB
  float* wsum     = (float*)carve((size_t)BATCH * KCODE * 4);          // 2 KB
  const size_t fragP = (size_t)ROWS_PROTO * D_DIM * 2;                 // 512 KB
  const size_t fragW = (size_t)D_DIM * D_DIM * 2;                      // 512 KB
  const size_t fragA = (size_t)ROWS_BIG * D_DIM * 2;                   // 32 MB
  unsigned short* afP_hi = (unsigned short*)carve(fragP);
  unsigned short* afP_lo = (unsigned short*)carve(fragP);
  unsigned short* bfP_hi = (unsigned short*)carve(fragW);
  unsigned short* bfP_lo = (unsigned short*)carve(fragW);
  unsigned short* bfU_hi = (unsigned short*)carve(fragW);
  unsigned short* bfU_lo = (unsigned short*)carve(fragW);
  unsigned short* afU_hi = (unsigned short*)carve(fragA);
  unsigned short* afU_lo = (unsigned short*)carve(fragA);

  // zero the accumulation buffers
  hipMemsetAsync(protoRaw, 0, (size_t)BATCH * KCODE * D_DIM * 4, stream);
  hipMemsetAsync(wsum, 0, (size_t)BATCH * KCODE * 4, stream);

  // 1) segmented weighted sum (128 KB + 256 B dynamic LDS per block)
  segsum_kernel<<<BATCH * 32, 256, (KCODE * D_DIM + KCODE) * 4, stream>>>(
      q, aidx, aw, cmask, protoRaw, wsum);

  // 2) pack both quaternion weight matrices into B fragments
  weight_pack_kernel<<<64, 256, 0, stream>>>(proto_w, (unsigned*)bfP_hi,
                                             (unsigned*)bfP_lo);
  weight_pack_kernel<<<64, 256, 0, stream>>>(upd_w, (unsigned*)bfU_hi,
                                             (unsigned*)bfU_lo);

  // 3) normalize proto + pack A fragments
  proto_pack_kernel<<<RT_PROTO, 256, 0, stream>>>(protoRaw, wsum,
                                                  (unsigned*)afP_hi,
                                                  (unsigned*)afP_lo);

  // 4) proto = proto @ Wp^T + bp   (writes second output region)
  qgemm_kernel<<<RT_PROTO / 2, 256, 2 * STAGE_BYTES, stream>>>(
      afP_hi, afP_lo, bfP_hi, bfP_lo, proto_b,
      nullptr, nullptr, nullptr, proto_out, 0);

  // 5) gather + Hamilton + weighted sum -> msg packed as A fragments
  msg_pack_kernel<<<RT_BIG, 256, 0, stream>>>(q, aidx, aw, proto_out,
                                              (unsigned*)afU_hi,
                                              (unsigned*)afU_lo);

  // 6) q_new = LN(q + msg @ Wu^T + bu)   (fused epilogue)
  qgemm_kernel<<<RT_BIG / 2, 256, 2 * STAGE_BYTES, stream>>>(
      afU_hi, afU_lo, bfU_hi, bfU_lo, upd_b, q, gamma, beta, qnew, 1);
}
